// GraphLinear_84198538871347
// MI455X (gfx1250) — matrix operand order repaired and checked by reading, not compiled, not run on hardware
//
#include <hip/hip_runtime.h>

typedef __attribute__((ext_vector_type(16))) _Float16 v16h;
typedef __attribute__((ext_vector_type(8)))  _Float16 v8h;
typedef __attribute__((ext_vector_type(4)))  _Float16 v4h;
typedef __attribute__((ext_vector_type(8)))  float    v8f;

#define NN   4096   // nodes
#define BB   8      // batch
#define HH   64     // hidden
#define FF   64     // in features
#define CTOT 512    // BB*HH fused columns

// LDS strides in halves. Multiples of 8 => every fragment run is 16B-aligned.
// 40 halves = 20 dwords: 5m mod 16 bijective => conflict-free b128 reads.
// 72 halves = 36 dwords: 9m mod 16 bijective => conflict-free b128 reads.
#define LDA3 40     // kernel3 A tile [128][32]
#define LDB3 40     // kernel3 B tile [64][32] ([col][k]!)
#define LDX2 72     // kernel2 x tile [128][64]
#define LDW2 72     // kernel2 W tile [64][64] ([h][f])
#define LDO2 136    // kernel2 output staging [64][128]

__device__ __forceinline__ v16h cat8(v8h lo, v8h hi) {
    return __builtin_shufflevector(lo, hi, 0, 1, 2, 3, 4, 5, 6, 7,
                                           8, 9, 10, 11, 12, 13, 14, 15);
}

// ---------------------------------------------------------------------------
// Kernel 1: inv_deg[i] = 1 / (1 + sum_j adj[i][j])
// ---------------------------------------------------------------------------
__global__ __launch_bounds__(256) void gcn_inv_deg(const float* __restrict__ adj,
                                                   float* __restrict__ inv_deg) {
    __shared__ float red[256];
    const int i = blockIdx.x;
    const float* row = adj + (size_t)i * NN;
    float s = 0.0f;
    for (int t = threadIdx.x; t < NN; t += 256) s += row[t];
    red[threadIdx.x] = s;
    __syncthreads();
    for (int off = 128; off > 0; off >>= 1) {
        if (threadIdx.x < off) red[threadIdx.x] += red[threadIdx.x + off];
        __syncthreads();
    }
    if (threadIdx.x == 0) inv_deg[i] = 1.0f / (red[0] + 1.0f);
}

// ---------------------------------------------------------------------------
// Kernel 2: embT[b*64+h][n] = f16( sum_f x[b,n,f] * W[h,f] + bias[h] )
// Output TRANSPOSED so kernel 3's B tiles are K-contiguous.
// 128 (b,n)-rows per block, 8 waves x 16 rows, WMMA f16.
// ---------------------------------------------------------------------------
__global__ __launch_bounds__(256) void gcn_emb(const float* __restrict__ x,
                                               const float* __restrict__ W,
                                               const float* __restrict__ bias,
                                               _Float16* __restrict__ embT) {
    __shared__ _Float16 ldsX[128 * LDX2];  // [row][f]
    __shared__ _Float16 ldsW[HH * LDW2];   // [h][f]  (B as [col][k])
    __shared__ _Float16 ldsO[HH * LDO2];   // [h][local row] staging

    const int tid  = threadIdx.x;
    const int lane = tid & 31;
    const int wave = tid >> 5;
    const int nlo  = lane & 15;
    const int hl   = lane >> 4;
    const int rowBase = blockIdx.x * 128;  // flattened b*NN + n (128 | NN)

    // Stage x tile [128][64] f32 -> f16
#pragma unroll
    for (int it = 0; it < 8; ++it) {
        int r  = it * 16 + (tid >> 4);
        int f4 = (tid & 15) * 4;
        float4 v = *(const float4*)(x + (size_t)(rowBase + r) * FF + f4);
        v4h h = { (_Float16)v.x, (_Float16)v.y, (_Float16)v.z, (_Float16)v.w };
        *(v4h*)&ldsX[r * LDX2 + f4] = h;
    }
    // Stage W [64][64] f32 -> f16 (row-major, already [h][f])
#pragma unroll
    for (int it = 0; it < 4; ++it) {
        int h  = it * 16 + (tid >> 4);
        int f4 = (tid & 15) * 4;
        float4 v = *(const float4*)(W + h * FF + f4);
        v4h hh = { (_Float16)v.x, (_Float16)v.y, (_Float16)v.z, (_Float16)v.w };
        *(v4h*)&ldsW[h * LDW2 + f4] = hh;
    }
    __syncthreads();

    v8f acc[4] = {};
#pragma unroll
    for (int step = 0; step < 2; ++step) {
        const int kb = step * 32;
        const int arow = (wave * 16 + nlo) * LDX2 + kb;
        v8h alo = *(const v8h*)&ldsX[arow + hl * 8];
        v8h ahi = *(const v8h*)&ldsX[arow + 16 + hl * 8];
        v16h a = cat8(alo, ahi);
#pragma unroll
        for (int t = 0; t < 4; ++t) {
            const int brow = (t * 16 + nlo) * LDW2 + kb + hl * 8;
            v8h blo = *(const v8h*)&ldsW[brow];
            v8h bhi = *(const v8h*)&ldsW[brow + 16];
            v16h bf = cat8(blo, bhi);
            acc[t] = __builtin_amdgcn_wmma_f32_16x16x32_f16(
                false, a, false, bf, (short)0, acc[t], false, false);
        }
    }

    // bias + stage transposed into LDS: ldsO[h][local row]
#pragma unroll
    for (int t = 0; t < 4; ++t) {
        int h = t * 16 + nlo;
        float bv = bias[h];
#pragma unroll
        for (int v = 0; v < 8; ++v) {
            int lr = wave * 16 + hl * 8 + v;
            ldsO[h * LDO2 + lr] = (_Float16)(acc[t][v] + bv);
        }
    }
    __syncthreads();

    // Cooperative coalesced store: embT[(b*64+h)][nBase..nBase+127]
    const int b     = rowBase >> 12;        // rowBase / NN
    const int nBase = rowBase & (NN - 1);
#pragma unroll
    for (int it = 0; it < 2; ++it) {
        int c   = it * 32 + (tid >> 3);     // h index 0..63
        int n16 = (tid & 7) * 16;
        const uint4* s = (const uint4*)&ldsO[c * LDO2 + n16];
        uint4* d = (uint4*)(embT + ((size_t)(b * HH + c)) * NN + nBase + n16);
        d[0] = s[0];
        d[1] = s[1];
    }
}

// ---------------------------------------------------------------------------
// Kernel 3: out[b,i,h] = inv_deg[i] * ( sum_j adj[i][j]*emb[j,bh] + emb[i,bh] )
// (adj + I) @ emb is split: the matmul uses raw adj; the identity term is
// added exactly (f32) in the epilogue -> no per-iteration diag compares.
// Tile 128(M) x 64(N), K-step 32. Double-buffered LDS, register-staged
// global loads pipelined against WMMA, one barrier per K step.
// ---------------------------------------------------------------------------
__global__ __launch_bounds__(256) void gcn_agg(const float* __restrict__ adj,
                                               const _Float16* __restrict__ embT,
                                               const float* __restrict__ inv_deg,
                                               float* __restrict__ out) {
    __shared__ _Float16 ldsA[2][128 * LDA3];  // adj tile [row][k] f16
    __shared__ _Float16 ldsB[2][64 * LDB3];   // emb tile [col][k] f16

    const int tid  = threadIdx.x;
    const int lane = tid & 31;
    const int wave = tid >> 5;
    const int nlo  = lane & 15;
    const int hl   = lane >> 4;
    const int rBase = blockIdx.x * 128;
    const int cBase = blockIdx.y * 64;

    // cooperative-load addressing
    const int r_ld = tid >> 3;          // 0..31 (adj: 4 iters of 32 rows)
    const int c4   = (tid & 7) * 4;     // adj k-offset
    const int cB   = tid >> 2;          // 0..63 (embT col)
    const int k8   = (tid & 3) * 8;     // embT k-offset

    float4 ar[4];
    uint4  er;

    auto load_tiles = [&](int kk) {
#pragma unroll
        for (int it = 0; it < 4; ++it)
            ar[it] = *(const float4*)(adj + (size_t)(rBase + it * 32 + r_ld) * NN + kk + c4);
        er = *(const uint4*)(embT + (size_t)(cBase + cB) * NN + kk + k8);
    };
    auto store_tiles = [&](int p) {
#pragma unroll
        for (int it = 0; it < 4; ++it) {
            int r = it * 32 + r_ld;
            v4h h = { (_Float16)ar[it].x, (_Float16)ar[it].y,
                      (_Float16)ar[it].z, (_Float16)ar[it].w };
            *(v4h*)&ldsA[p][r * LDA3 + c4] = h;
        }
        *(uint4*)&ldsB[p][cB * LDB3 + k8] = er;
    };

    v8f acc[4] = {};

    load_tiles(0);
    store_tiles(0);
    __syncthreads();

    for (int kk = 0; kk < NN; kk += 32) {
        const int p = (kk >> 5) & 1;
        const bool more = (kk + 32) < NN;
        if (more) load_tiles(kk + 32);   // in flight while WMMAs run

        const int arow = (wave * 16 + nlo) * LDA3;
        v8h alo = *(const v8h*)&ldsA[p][arow + hl * 8];
        v8h ahi = *(const v8h*)&ldsA[p][arow + 16 + hl * 8];
        v16h a = cat8(alo, ahi);
#pragma unroll
        for (int t = 0; t < 4; ++t) {
            const int brow = (t * 16 + nlo) * LDB3 + hl * 8;
            v8h blo = *(const v8h*)&ldsB[p][brow];
            v8h bhi = *(const v8h*)&ldsB[p][brow + 16];
            v16h bf = cat8(blo, bhi);
            acc[t] = __builtin_amdgcn_wmma_f32_16x16x32_f16(
                false, a, false, bf, (short)0, acc[t], false, false);
        }

        if (more) store_tiles(p ^ 1);
        __syncthreads();
    }

    // Epilogue: add identity term (self embedding), scale by 1/deg, scatter.
#pragma unroll
    for (int v = 0; v < 8; ++v) {
        int i = rBase + wave * 16 + hl * 8 + v;
        float d = inv_deg[i];
#pragma unroll
        for (int t = 0; t < 4; ++t) {
            int c = cBase + t * 16 + nlo;
            int b = c >> 6;
            int h = c & 63;
            float self = (float)embT[(size_t)c * NN + i];  // I @ emb term
            out[((size_t)b * NN + i) * HH + h] = (acc[t][v] + self) * d;
        }
    }
}

// ---------------------------------------------------------------------------
extern "C" void kernel_launch(void* const* d_in, const int* in_sizes, int n_in,
                              void* d_out, int out_size, void* d_ws, size_t ws_size,
                              hipStream_t stream) {
    const float* x    = (const float*)d_in[0];  // [8, 4096, 64]
    const float* adj  = (const float*)d_in[1];  // [4096, 4096]
    const float* W    = (const float*)d_in[2];  // [64, 64]
    const float* bias = (const float*)d_in[3];  // [64]
    float* out = (float*)d_out;                 // [8, 4096, 64]

    float*    inv_deg = (float*)d_ws;                                  // 16 KB
    _Float16* embT    = (_Float16*)((char*)d_ws + NN * sizeof(float)); // 4 MB, [512][4096]

    gcn_inv_deg<<<NN, 256, 0, stream>>>(adj, inv_deg);
    gcn_emb<<<(BB * NN) / 128, 256, 0, stream>>>(x, W, bias, embT);
    dim3 grid(NN / 128, CTOT / 64);  // 32 x 8
    gcn_agg<<<grid, 256, 0, stream>>>(adj, embT, inv_deg, out);
}
